// Block_38474317038049
// MI455X (gfx1250) — matrix operand order repaired and checked by reading, not compiled, not run on hardware
//
#include <hip/hip_runtime.h>
#include <hip/hip_bf16.h>

#define C_DIM   1024
#define T_DIM   1024
#define B_DIM   2
#define H_DIM   16
#define HS_DIM  64
#define L_KV    1024
#define KT_FULL (L_KV + T_DIM)   // 2048 keys total

typedef __attribute__((ext_vector_type(16))) __bf16 v16bf;
typedef __attribute__((ext_vector_type(8)))  float  v8f;
typedef __attribute__((ext_vector_type(4)))  unsigned int u32x4;
typedef __attribute__((ext_vector_type(8)))  int i32x8;
typedef __attribute__((ext_vector_type(4)))  int i32x4;

union Frag16 { v16bf v; uint4 u[2]; };

__device__ __forceinline__ v8f wmma_bf16(const Frag16& a, const Frag16& b, v8f c) {
  return __builtin_amdgcn_wmma_f32_16x16x32_bf16(false, a.v, false, b.v,
                                                 (short)0, c, false, false);
}

// -------------------------------------------------------------- TDM 2-D tile
// Loads a tile of tile1 rows x tile0 (2-byte) elements from a row-major
// tensor with row stride `stride0` elements into LDS at byte offset lds_off.
__device__ __forceinline__ void tdm_load_tile2d(const void* gptr, unsigned lds_off,
                                                unsigned td0, unsigned tile0,
                                                unsigned tile1,
                                                unsigned long long stride0) {
  unsigned long long ga = (unsigned long long)(uintptr_t)gptr;
  u32x4 g0;
  g0.x = 1u;                                    // count=1, user mode
  g0.y = lds_off;                               // LDS byte address
  g0.z = (unsigned)ga;                          // global_addr[31:0]
  g0.w = (unsigned)(ga >> 32) | (2u << 30);     // global_addr[56:32] | type=2
  const unsigned td1 = 0x100000u;               // big tensor_dim1 (no OOB)
  i32x8 g1;
  g1[0] = 0x00010000;                           // data_size=1 (2 bytes)
  g1[1] = (int)((td0 & 0xffffu) << 16);         // tensor_dim0[15:0] @ bits 63:48
  g1[2] = (int)((td0 >> 16) | ((td1 & 0xffffu) << 16));
  g1[3] = (int)((td1 >> 16) | (tile0 << 16));   // tile_dim0 @ bits 127:112
  g1[4] = (int)tile1;                           // tile_dim1, tile_dim2=0
  g1[5] = (int)(unsigned)stride0;               // tensor_dim0_stride[31:0]
  g1[6] = (int)(unsigned)(stride0 >> 32);       // stride[47:32], dim1_stride lo=0
  g1[7] = 0;
  i32x4 gz4 = {0, 0, 0, 0};
  i32x8 gz8 = {0, 0, 0, 0, 0, 0, 0, 0};
  __builtin_amdgcn_tensor_load_to_lds(g0, g1, gz4, gz4, gz8, 0);
}

// ---------------------------------------------------------------- conversions
// W [K,N] f32 -> Wt [N,K] bf16 (tiled transpose; one-shot, off critical path)
__global__ __launch_bounds__(256) void transpose_f32_to_bf16_kernel(
    const float* __restrict__ in, __bf16* __restrict__ out, int K, int N) {
  __shared__ float tile[32][33];
  const int k0 = blockIdx.y * 32, n0 = blockIdx.x * 32;
  const int tx = threadIdx.x & 31, ty = threadIdx.x >> 5;   // 32 x 8
#pragma unroll
  for (int j = 0; j < 32; j += 8)
    tile[ty + j][tx] = in[(size_t)(k0 + ty + j) * N + n0 + tx];
  __syncthreads();
#pragma unroll
  for (int j = 0; j < 32; j += 8)
    out[(size_t)(n0 + ty + j) * K + k0 + tx] = (__bf16)tile[tx][ty + j];
}

// k cache [B,H,L,HS] f32 -> rows [0,L) of [B,H,L+T,HS] bf16
__global__ void kcache_to_full_kernel(const float* __restrict__ in,
                                      __bf16* __restrict__ out, int n) {
  int i = blockIdx.x * 256 + threadIdx.x;
  if (i < n) {
    int hs = i & 63;
    int l  = (i >> 6) & 1023;
    int bh = i >> 16;                       // HS*L = 65536
    out[((size_t)bh * KT_FULL + l) * HS_DIM + hs] = (__bf16)in[i];
  }
}

// v cache [B,H,L,HS] f32 -> cols [0,L) of transposed [B,H,HS,L+T] bf16
__global__ void vcache_to_full_t_kernel(const float* __restrict__ in,
                                        __bf16* __restrict__ out, int n) {
  int i = blockIdx.x * 256 + threadIdx.x;
  if (i < n) {
    int hs = i & 63;
    int l  = (i >> 6) & 1023;
    int bh = i >> 16;
    out[((size_t)bh * HS_DIM + hs) * KT_FULL + l] = (__bf16)in[i];
  }
}

// ---------------------------------------------------------------- layernorm
__global__ __launch_bounds__(256) void layernorm_bf16_kernel(
    const float* __restrict__ x, const float* __restrict__ w,
    const float* __restrict__ b, __bf16* __restrict__ out) {
  const int row = blockIdx.x, tid = threadIdx.x;
  const int lane = tid & 31, wid = tid >> 5;
  const float* xr = x + (size_t)row * C_DIM;
  float v[4]; float s = 0.f, s2 = 0.f;
#pragma unroll
  for (int j = 0; j < 4; ++j) { v[j] = xr[tid + j * 256]; s += v[j]; s2 += v[j] * v[j]; }
#pragma unroll
  for (int m = 16; m >= 1; m >>= 1) { s += __shfl_xor(s, m); s2 += __shfl_xor(s2, m); }
  __shared__ float ps[8], ps2[8];
  if (lane == 0) { ps[wid] = s; ps2[wid] = s2; }
  __syncthreads();
  float tot = 0.f, tot2 = 0.f;
#pragma unroll
  for (int i = 0; i < 8; ++i) { tot += ps[i]; tot2 += ps2[i]; }
  const float mu = tot * (1.f / C_DIM);
  const float var = tot2 * (1.f / C_DIM) - mu * mu;
  const float rs = rsqrtf(var + 1e-5f);
#pragma unroll
  for (int j = 0; j < 4; ++j) {
    int c = tid + j * 256;
    out[(size_t)row * C_DIM + c] = (__bf16)((v[j] - mu) * rs * w[c] + b[c]);
  }
}

// ---------------------------------------------------------------- GEMM (WMMA)
// A [M,K] bf16, Bt [N,K] bf16 (pre-transposed weights).
// Tiles staged by TDM with double buffering (DMA overlaps WMMA stage).
enum { EPI_QKV = 0, EPI_BIAS_RES_F32 = 1, EPI_BIAS_GELU_BF16 = 2 };

template <int EPI>
__global__ __launch_bounds__(256) void gemm_bf16_wmma(
    const __bf16* __restrict__ A, const __bf16* __restrict__ Bt,
    const float* __restrict__ bias, const float* __restrict__ resid,
    float* __restrict__ outF, __bf16* __restrict__ outB,
    __bf16* __restrict__ qo, __bf16* __restrict__ ko, __bf16* __restrict__ vo,
    int M, int N, int K) {
  __shared__ __align__(16) __bf16 As[2][128 * 32];   // [m][k] double buffered
  __shared__ __align__(16) __bf16 Bs[2][128 * 32];   // [n][k] double buffered
  const int tid = threadIdx.x;
  const int lane = tid & 31, wid = tid >> 5;
  const int half = lane >> 4, r = lane & 15;
  const int waveM = wid >> 2, waveN = wid & 3;       // 2 x 4 wave grid
  const int m0 = blockIdx.y * 128, n0 = blockIdx.x * 128;

  v8f acc[4][2];
#pragma unroll
  for (int i = 0; i < 4; ++i)
#pragma unroll
    for (int j = 0; j < 2; ++j)
#pragma unroll
      for (int e = 0; e < 8; ++e) acc[i][j][e] = 0.f;

  // prologue: kick off tiles for k-step 0
  if (wid == 0) {
    tdm_load_tile2d(A  + (size_t)m0 * K, (unsigned)(uintptr_t)(void*)&As[0][0],
                    (unsigned)K, 32, 128, (unsigned long long)K);
    tdm_load_tile2d(Bt + (size_t)n0 * K, (unsigned)(uintptr_t)(void*)&Bs[0][0],
                    (unsigned)K, 32, 128, (unsigned long long)K);
  }

  const int nk = K >> 5;
  for (int i = 0; i < nk; ++i) {
    const int cur = i & 1;
    __syncthreads();   // all waves done reading buffer (cur^1) from iter i-1
    if (wid == 0) {
      __builtin_amdgcn_s_wait_tensorcnt(0);   // current tiles have landed
      if (i + 1 < nk) {                        // prefetch next tiles via TDM
        const int kt = (i + 1) << 5;
        tdm_load_tile2d(A  + (size_t)m0 * K + kt,
                        (unsigned)(uintptr_t)(void*)&As[cur ^ 1][0],
                        (unsigned)K, 32, 128, (unsigned long long)K);
        tdm_load_tile2d(Bt + (size_t)n0 * K + kt,
                        (unsigned)(uintptr_t)(void*)&Bs[cur ^ 1][0],
                        (unsigned)K, 32, 128, (unsigned long long)K);
      }
    }
    __syncthreads();   // current tiles visible to all waves

    Frag16 af[4], bfr[2];
#pragma unroll
    for (int mf = 0; mf < 4; ++mf) {
      int row = waveM * 64 + mf * 16 + r;
      af[mf].u[0] = *(const uint4*)&As[cur][row * 32 + 8 * half];
      af[mf].u[1] = *(const uint4*)&As[cur][row * 32 + 16 + 8 * half];
    }
#pragma unroll
    for (int nf = 0; nf < 2; ++nf) {
      int col = waveN * 32 + nf * 16 + r;
      bfr[nf].u[0] = *(const uint4*)&Bs[cur][col * 32 + 8 * half];
      bfr[nf].u[1] = *(const uint4*)&Bs[cur][col * 32 + 16 + 8 * half];
    }
#pragma unroll
    for (int mf = 0; mf < 4; ++mf)
#pragma unroll
      for (int nf = 0; nf < 2; ++nf)
        acc[mf][nf] = wmma_bf16(af[mf], bfr[nf], acc[mf][nf]);
  }

  // epilogue
#pragma unroll
  for (int mf = 0; mf < 4; ++mf)
#pragma unroll
    for (int nf = 0; nf < 2; ++nf) {
      const int gcol = n0 + waveN * 32 + nf * 16 + r;
      const float bv = bias ? bias[gcol] : 0.f;
#pragma unroll
      for (int e = 0; e < 8; ++e) {
        const int grow = m0 + waveM * 64 + mf * 16 + e + 8 * half;
        const float val = acc[mf][nf][e] + bv;
        if (EPI == EPI_QKV) {
          int b = grow >> 10, t = grow & 1023;         // T = 1024
          int sel = gcol >> 10, cc = gcol & 1023;      // C = 1024
          int h = cc >> 6, hs = cc & 63;               // HS = 64
          size_t hd = (size_t)(b * H_DIM + h);
          if (sel == 0)
            qo[(hd * T_DIM + t) * HS_DIM + hs] = (__bf16)(val * 0.125f); // 1/sqrt(64)
          else if (sel == 1)
            ko[(hd * KT_FULL + L_KV + t) * HS_DIM + hs] = (__bf16)val;
          else  // V stored transposed: [B,H,HS,L+T]
            vo[(hd * HS_DIM + hs) * KT_FULL + L_KV + t] = (__bf16)val;
        } else if (EPI == EPI_BIAS_RES_F32) {
          size_t idx = (size_t)grow * N + gcol;
          outF[idx] = val + resid[idx];
        } else {
          float g = 0.5f * val * (1.0f + erff(val * 0.70710678f));
          outB[(size_t)grow * N + gcol] = (__bf16)g;
        }
      }
    }
}

// ---------------------------------------------------------------- attention
// One wave per PAIR of adjacent 16-query tiles of one (b,h): every K/V
// fragment load is shared by both tiles (2x arithmetic intensity, halves
// the L2 K/V re-read amplification). Flash-style online softmax.
// K: [B,H,L+T,HS] (contraction dim hs contiguous).
// V: [B,H,HS,L+T] (contraction dim key contiguous) -> no LDS staging needed.
__global__ __launch_bounds__(128) void attn_fa_wmma(
    const __bf16* __restrict__ q, const __bf16* __restrict__ kf,
    const __bf16* __restrict__ vt, __bf16* __restrict__ yb) {
  __shared__ __align__(16) __bf16 Pa[4][2][16 * 32];  // per-wave, per-tile P
  const int wid = threadIdx.x >> 5, lane = threadIdx.x & 31;
  const int half = lane >> 4, r = lane & 15;
  const int task = blockIdx.x * 4 + wid;     // 1024 tasks total
  const int pair = task & 31;                // 32 tile-pairs per (b,h)
  const int bh = task >> 5;                  // 0..31
  const int qt0 = pair * 2;                  // tiles qt0, qt0+1
  const __bf16* qp = q + ((size_t)bh * T_DIM + qt0 * 16) * HS_DIM;
  const __bf16* kp = kf + (size_t)bh * KT_FULL * HS_DIM;
  const __bf16* vp = vt + (size_t)bh * HS_DIM * KT_FULL;

  // Q as A-operands: [tile][k-chunk]; contraction dim contiguous in memory.
  Frag16 qa[2][2];
#pragma unroll
  for (int tl = 0; tl < 2; ++tl)
#pragma unroll
    for (int kk = 0; kk < 2; ++kk) {
      const __bf16* qr = qp + (size_t)(tl * 16 + r) * HS_DIM + kk * 32;
      qa[tl][kk].u[0] = *(const uint4*)(qr + 16 * half);
      qa[tl][kk].u[1] = *(const uint4*)(qr + 16 * half + 8);
    }

  float Mr[2][8], Sr[2][8];
  v8f O[2][4];
#pragma unroll
  for (int tl = 0; tl < 2; ++tl) {
#pragma unroll
    for (int e = 0; e < 8; ++e) { Mr[tl][e] = -1e30f; Sr[tl][e] = 0.f; }
#pragma unroll
    for (int nf = 0; nf < 4; ++nf)
#pragma unroll
      for (int e = 0; e < 8; ++e) O[tl][nf][e] = 0.f;
  }

  const int kend = L_KV + (qt0 + 2) * 16;    // multiple of 32
  for (int kb = 0; kb < kend; kb += 32) {
    // ---- S = Q * K^T for 32 keys; K fragments shared by both q-tiles
    v8f s[2][2];                             // [tile][sub]
#pragma unroll
    for (int sub = 0; sub < 2; ++sub) {
      const __bf16* krow = kp + (size_t)(kb + sub * 16 + r) * HS_DIM;
      Frag16 k0, k1;
      k0.u[0] = *(const uint4*)(krow + 16 * half);
      k0.u[1] = *(const uint4*)(krow + 16 * half + 8);
      k1.u[0] = *(const uint4*)(krow + 32 + 16 * half);
      k1.u[1] = *(const uint4*)(krow + 32 + 16 * half + 8);
#pragma unroll
      for (int tl = 0; tl < 2; ++tl) {
        v8f z;
#pragma unroll
        for (int e = 0; e < 8; ++e) z[e] = 0.f;
        z = wmma_bf16(qa[tl][0], k0, z);
        s[tl][sub] = wmma_bf16(qa[tl][1], k1, z);
      }
    }
    if (kb + 32 < kend)
      __builtin_prefetch(kp + (size_t)(kb + 32 + lane) * HS_DIM, 0, 3);

    // ---- causal mask + online softmax per tile
#pragma unroll
    for (int tl = 0; tl < 2; ++tl) {
      const int qbase = L_KV + (qt0 + tl) * 16 + 8 * half;
#pragma unroll
      for (int sub = 0; sub < 2; ++sub)
#pragma unroll
        for (int e = 0; e < 8; ++e)
          if (kb + sub * 16 + r > qbase + e) s[tl][sub][e] = -1e30f;

      __bf16* Paw = Pa[wid][tl];
#pragma unroll
      for (int e = 0; e < 8; ++e) {
        float rmax = fmaxf(s[tl][0][e], s[tl][1][e]);
        rmax = fmaxf(rmax, __shfl_xor(rmax, 1));
        rmax = fmaxf(rmax, __shfl_xor(rmax, 2));
        rmax = fmaxf(rmax, __shfl_xor(rmax, 4));
        rmax = fmaxf(rmax, __shfl_xor(rmax, 8));
        float nm = fmaxf(Mr[tl][e], rmax);
        float alpha = __expf(Mr[tl][e] - nm);
        Mr[tl][e] = nm;
        float p0 = __expf(s[tl][0][e] - nm);
        float p1 = __expf(s[tl][1][e] - nm);
        float psum = p0 + p1;
        psum += __shfl_xor(psum, 1);
        psum += __shfl_xor(psum, 2);
        psum += __shfl_xor(psum, 4);
        psum += __shfl_xor(psum, 8);
        Sr[tl][e] = Sr[tl][e] * alpha + psum;
#pragma unroll
        for (int nf = 0; nf < 4; ++nf) O[tl][nf][e] *= alpha;
        // C-layout -> row-major P tile in LDS (row m = e + 8*half, col = key)
        Paw[(e + 8 * half) * 32 + r] = (__bf16)p0;
        Paw[(e + 8 * half) * 32 + 16 + r] = (__bf16)p1;
      }
    }

    // ---- O += P * V : V fragments shared by both q-tiles
    Frag16 pa[2];
#pragma unroll
    for (int tl = 0; tl < 2; ++tl) {
      pa[tl].u[0] = *(const uint4*)&Pa[wid][tl][r * 32 + 16 * half];
      pa[tl].u[1] = *(const uint4*)&Pa[wid][tl][r * 32 + 16 * half + 8];
    }
#pragma unroll
    for (int nf = 0; nf < 4; ++nf) {
      Frag16 vb;
      const __bf16* vrow = vp + (size_t)(nf * 16 + r) * KT_FULL + kb + 16 * half;
      vb.u[0] = *(const uint4*)vrow;
      vb.u[1] = *(const uint4*)(vrow + 8);
#pragma unroll
      for (int tl = 0; tl < 2; ++tl)
        O[tl][nf] = wmma_bf16(pa[tl], vb, O[tl][nf]);
    }
  }

  // ---- finalize: y[b, t, h*64 + hs] = O / Sr
  const int b = bh >> 4, h = bh & 15;
#pragma unroll
  for (int tl = 0; tl < 2; ++tl)
#pragma unroll
    for (int nf = 0; nf < 4; ++nf)
#pragma unroll
      for (int e = 0; e < 8; ++e) {
        int t = (qt0 + tl) * 16 + e + 8 * half;
        int col = h * HS_DIM + nf * 16 + r;
        yb[((size_t)b * T_DIM + t) * C_DIM + col] = (__bf16)(O[tl][nf][e] / Sr[tl][e]);
      }
}

// ---------------------------------------------------------------- launch
extern "C" void kernel_launch(void* const* d_in, const int* in_sizes, int n_in,
                              void* d_out, int out_size, void* d_ws, size_t ws_size,
                              hipStream_t stream) {
  const float* x      = (const float*)d_in[0];
  const float* kcache = (const float*)d_in[1];
  const float* vcache = (const float*)d_in[2];
  const float* ln1w   = (const float*)d_in[3];
  const float* ln1b   = (const float*)d_in[4];
  const float* Wqkv   = (const float*)d_in[5];
  const float* bqkv   = (const float*)d_in[6];
  const float* Wproj  = (const float*)d_in[7];
  const float* bproj  = (const float*)d_in[8];
  const float* ln2w   = (const float*)d_in[9];
  const float* ln2b   = (const float*)d_in[10];
  const float* Wfc    = (const float*)d_in[11];
  const float* bfc    = (const float*)d_in[12];
  const float* Wfc2   = (const float*)d_in[13];
  const float* bfc2   = (const float*)d_in[14];

  char* ws = (char*)d_ws;
  size_t off = 0;
  auto take = [&](size_t bytes) -> void* {
    void* p = ws + off;
    off = (off + bytes + 255) & ~(size_t)255;
    return p;
  };
  const int MT = B_DIM * T_DIM;  // 2048 rows
  __bf16* hb     = (__bf16*)take((size_t)MT * C_DIM * 2);
  __bf16* Wqkvt  = (__bf16*)take((size_t)C_DIM * 3 * C_DIM * 2);   // [3C, C]
  __bf16* Wprojt = (__bf16*)take((size_t)C_DIM * C_DIM * 2);       // [C, C]
  __bf16* Wfct   = (__bf16*)take((size_t)C_DIM * 4 * C_DIM * 2);   // [4C, C]
  __bf16* Wfc2t  = (__bf16*)take((size_t)4 * C_DIM * C_DIM * 2);   // [C, 4C]
  __bf16* qb     = (__bf16*)take((size_t)MT * C_DIM * 2);
  __bf16* kfull  = (__bf16*)take((size_t)B_DIM * H_DIM * KT_FULL * HS_DIM * 2);
  __bf16* vfullt = (__bf16*)take((size_t)B_DIM * H_DIM * HS_DIM * KT_FULL * 2);
  __bf16* ybf    = (__bf16*)take((size_t)MT * C_DIM * 2);
  float*  x1     = (float*) take((size_t)MT * C_DIM * 4);
  __bf16* h2b    = (__bf16*)take((size_t)MT * C_DIM * 2);
  __bf16* gb     = (__bf16*)take((size_t)MT * 4 * C_DIM * 2);

  // weights: [K,N] f32 -> [N,K] bf16
  transpose_f32_to_bf16_kernel<<<dim3(3 * C_DIM / 32, C_DIM / 32), 256, 0, stream>>>(
      Wqkv, Wqkvt, C_DIM, 3 * C_DIM);
  transpose_f32_to_bf16_kernel<<<dim3(C_DIM / 32, C_DIM / 32), 256, 0, stream>>>(
      Wproj, Wprojt, C_DIM, C_DIM);
  transpose_f32_to_bf16_kernel<<<dim3(4 * C_DIM / 32, C_DIM / 32), 256, 0, stream>>>(
      Wfc, Wfct, C_DIM, 4 * C_DIM);
  transpose_f32_to_bf16_kernel<<<dim3(C_DIM / 32, 4 * C_DIM / 32), 256, 0, stream>>>(
      Wfc2, Wfc2t, 4 * C_DIM, C_DIM);

  int n = B_DIM * H_DIM * L_KV * HS_DIM;
  kcache_to_full_kernel<<<(n + 255) / 256, 256, 0, stream>>>(kcache, kfull, n);
  vcache_to_full_t_kernel<<<(n + 255) / 256, 256, 0, stream>>>(vcache, vfullt, n);

  layernorm_bf16_kernel<<<MT, 256, 0, stream>>>(x, ln1w, ln1b, hb);

  gemm_bf16_wmma<EPI_QKV><<<dim3(3 * C_DIM / 128, MT / 128), 256, 0, stream>>>(
      hb, Wqkvt, bqkv, nullptr, nullptr, nullptr, qb, kfull, vfullt,
      MT, 3 * C_DIM, C_DIM);

  attn_fa_wmma<<<(B_DIM * H_DIM * (T_DIM / 16) / 2) / 4, 128, 0, stream>>>(
      qb, kfull, vfullt, ybf);

  gemm_bf16_wmma<EPI_BIAS_RES_F32><<<dim3(C_DIM / 128, MT / 128), 256, 0, stream>>>(
      ybf, Wprojt, bproj, x, x1, nullptr, nullptr, nullptr, nullptr,
      MT, C_DIM, C_DIM);

  layernorm_bf16_kernel<<<MT, 256, 0, stream>>>(x1, ln2w, ln2b, h2b);

  gemm_bf16_wmma<EPI_BIAS_GELU_BF16><<<dim3(4 * C_DIM / 128, MT / 128), 256, 0, stream>>>(
      h2b, Wfct, bfc, nullptr, nullptr, gb, nullptr, nullptr, nullptr,
      MT, 4 * C_DIM, C_DIM);

  gemm_bf16_wmma<EPI_BIAS_RES_F32><<<dim3(C_DIM / 128, MT / 128), 256, 0, stream>>>(
      gb, Wfc2t, bfc2, x1, (float*)d_out, nullptr, nullptr, nullptr, nullptr,
      MT, C_DIM, 4 * C_DIM);
}
